// MultiScaleRetention_82583631167651
// MI455X (gfx1250) — matrix operand order repaired and checked by reading, not compiled, not run on hardware
//
#include <hip/hip_runtime.h>
#include <hip/hip_bf16.h>

#define S_LEN    2048
#define HID      1024
#define NHEADS   16
#define DHEAD    64
#define BM       64     // query rows per workgroup
#define BK       64     // key rows per iteration
#define NTHREADS 128    // 4 waves (wave32)
#define LSTR     72     // padded LDS row stride in bf16 elems (144 bytes)

typedef __attribute__((ext_vector_type(16))) __bf16 v16bf;
typedef __attribute__((ext_vector_type(8)))  float  v8f;

union Frag16 {
    v16bf bf;
    uint4 q[2];
};

__device__ __forceinline__ unsigned short f2bf(float f) {
    unsigned int u = __float_as_uint(f);
    unsigned int r = u + 0x7FFFu + ((u >> 16) & 1u);
    return (unsigned short)(r >> 16);
}
__device__ __forceinline__ unsigned int pack2(float lo, float hi) {
    return (unsigned int)f2bf(lo) | ((unsigned int)f2bf(hi) << 16);
}

// xpos rotary embedding with per-dimension scale; sgn=+1 => scale, sgn=-1 => 1/scale
__device__ __forceinline__ void xpos4(float4& x, int pos, int d0, float sgn) {
    const float L2_10000 = 13.287712379549449f;
    float fpos = (float)pos;
#pragma unroll
    for (int p = 0; p < 2; ++p) {
        int j = (d0 >> 1) + p;
        float bs  = (2.0f * (float)j + 25.6f) * (1.0f / 89.6f);
        float sc  = exp2f(log2f(bs) * (fpos * (1.0f / 512.0f)) * sgn);
        float inf = exp2f(-((float)j * (1.0f / 32.0f)) * L2_10000);
        float ang = fpos * inf;
        float sn = __sinf(ang) * sc;
        float cs = __cosf(ang) * sc;
        float xe = p ? x.z : x.x;
        float xo = p ? x.w : x.y;
        float oe = xe * cs - xo * sn;
        float oo = xo * cs + xe * sn;
        if (p) { x.z = oe; x.w = oo; } else { x.x = oe; x.y = oo; }
    }
}

__device__ __forceinline__ float head_log2gamma(int h) {
    const float lga = -3.4657359027997265f;  // log(1/32)
    const float lgb = -6.2383246250395075f;  // log(1/512)
    float gamma = 1.0f - __expf(lga + (float)h * ((lgb - lga) * (1.0f / 15.0f)));
    return log2f(gamma);
}

// ---- CDNA5 async global->LDS copy (ASYNCcnt-tracked DMA) ----
__device__ __forceinline__ void async_b128(unsigned lds_addr, unsigned long long gaddr) {
    asm volatile("global_load_async_to_lds_b128 %0, %1, off"
                 :: "v"(lds_addr), "v"(gaddr) : "memory");
}
__device__ __forceinline__ void wait_async0() {
    asm volatile("s_wait_asynccnt 0" ::: "memory");
}

// =====================================================================
// Pass 1: xpos(Q) -> bf16; xpos(K) * gamma^{-(s mod 64)} -> bf16 (decay
// column factor baked in, per head); V -> bf16 transposed (d-major).
// =====================================================================
__global__ __launch_bounds__(NTHREADS)
void retention_prep(const float* __restrict__ Q, const float* __restrict__ K,
                    const float* __restrict__ V,
                    unsigned short* __restrict__ wsQ, unsigned short* __restrict__ wsK,
                    unsigned short* __restrict__ wsVT) {
    __shared__ unsigned short vts[DHEAD * LSTR];

    const int tid = threadIdx.x;
    const int sb  = blockIdx.x & 31;
    const int h   = (blockIdx.x >> 5) & 15;
    const int b   = blockIdx.x >> 9;
    const int s0  = sb * 64;

    const float l2g = head_log2gamma(h);
    const size_t bh_off = (size_t)b * S_LEN * HID + (size_t)h * DHEAD;

#pragma unroll
    for (int it = 0; it < 8; ++it) {
        int e  = tid + it * NTHREADS;
        int r  = e >> 4;
        int d0 = (e & 15) << 2;
        int srow = s0 + r;
        size_t off = bh_off + (size_t)srow * HID + d0;

        float4 x = *(const float4*)(Q + off);
        xpos4(x, srow, d0, 1.0f);
        *(uint2*)&wsQ[off] = make_uint2(pack2(x.x, x.y), pack2(x.z, x.w));

        float4 k = *(const float4*)(K + off);
        xpos4(k, srow, d0, -1.0f);
        float gd = exp2f(-l2g * (float)(srow & 63));   // gamma^{-(s mod 64)} <= gamma^-63
        k.x *= gd; k.y *= gd; k.z *= gd; k.w *= gd;
        *(uint2*)&wsK[off] = make_uint2(pack2(k.x, k.y), pack2(k.z, k.w));

        float4 y = *(const float4*)(V + off);
        vts[(d0 + 0) * LSTR + r] = f2bf(y.x);
        vts[(d0 + 1) * LSTR + r] = f2bf(y.y);
        vts[(d0 + 2) * LSTR + r] = f2bf(y.z);
        vts[(d0 + 3) * LSTR + r] = f2bf(y.w);
    }
    __syncthreads();

    const size_t vbase = ((size_t)(b * NHEADS + h) * DHEAD) * S_LEN + s0;
#pragma unroll
    for (int it = 0; it < 16; ++it) {
        int e = tid + it * NTHREADS;
        int d = e >> 5;
        int c = (e & 31) << 1;
        *(unsigned int*)&wsVT[vbase + (size_t)d * S_LEN + c] =
            *(const unsigned int*)&vts[d * LSTR + c];
    }
}

// =====================================================================
// One key-block step: GEMM1 (+row decay, optional diag mask) -> GEMM2.
// DIAG instantiation is peeled so the steady-state loop is mask-free.
// =====================================================================
template<bool DIAG>
__device__ __forceinline__ void tile_step(const unsigned short* ksb, const unsigned short* vtsb,
                                          unsigned short* sw, const Frag16 aq[2],
                                          v8f acc[4], const float fv[8],
                                          int w, int ln, int hi) {
    __bf16* swb = (__bf16*)sw;

    // ---- GEMM1: scores (with baked column decay), scale rows by fv ----
#pragma unroll
    for (int t = 0; t < 4; ++t) {
        if (DIAG && t > w) {         // fully masked tile: store zeros
#pragma unroll
            for (int v2 = 0; v2 < 8; ++v2)
                sw[(v2 + 8 * hi) * LSTR + t * 16 + ln] = 0;
            continue;
        }
        Frag16 bk0, bk1;
        const unsigned short* kr = &ksb[(t * 16 + ln) * LSTR];
        bk0.q[0] = *(const uint4*)(kr + hi * 8);
        bk0.q[1] = *(const uint4*)(kr + 16 + hi * 8);
        bk1.q[0] = *(const uint4*)(kr + 32 + hi * 8);
        bk1.q[1] = *(const uint4*)(kr + 48 + hi * 8);

        v8f c = {};
        c = __builtin_amdgcn_wmma_f32_16x16x32_bf16(false, aq[0].bf, false, bk0.bf,
                                                    (short)0, c, false, false);
        c = __builtin_amdgcn_wmma_f32_16x16x32_bf16(false, aq[1].bf, false, bk1.bf,
                                                    (short)0, c, false, false);

        if (DIAG && t == w) {        // partially masked tile: n < m -> 0
            int e0 = 8 * hi - ln;    // (w-t)*16 + v + 8*hi - ln with t==w
#pragma unroll
            for (int v2 = 0; v2 < 8; ++v2) {
                float val = ((e0 + v2) < 0) ? 0.0f : c[v2] * fv[v2];
                swb[(v2 + 8 * hi) * LSTR + t * 16 + ln] = (__bf16)val;
            }
        } else {
#pragma unroll
            for (int v2 = 0; v2 < 8; ++v2)
                swb[(v2 + 8 * hi) * LSTR + t * 16 + ln] = (__bf16)(c[v2] * fv[v2]);
        }
    }

    // ---- GEMM2: out += scores . v ----
    Frag16 a2[2];
#pragma unroll
    for (int mh = 0; mh < 2; ++mh) {
        const unsigned short* sr = &sw[ln * LSTR];
        a2[mh].q[0] = *(const uint4*)(sr + mh * 32 + hi * 8);
        a2[mh].q[1] = *(const uint4*)(sr + mh * 32 + 16 + hi * 8);
    }
#pragma unroll
    for (int dt = 0; dt < 4; ++dt) {
#pragma unroll
        for (int mh = 0; mh < 2; ++mh) {
            Frag16 bv;
            const unsigned short* vr = &vtsb[(dt * 16 + ln) * LSTR];
            bv.q[0] = *(const uint4*)(vr + mh * 32 + hi * 8);
            bv.q[1] = *(const uint4*)(vr + mh * 32 + 16 + hi * 8);
            acc[dt] = __builtin_amdgcn_wmma_f32_16x16x32_bf16(false, a2[mh].bf, false, bv.bf,
                                                              (short)0, acc[dt], false, false);
        }
    }
}

// =====================================================================
// Pass 2: retention GEMMs on bf16 with WMMA, async double-buffered tiles.
// =====================================================================
__global__ __launch_bounds__(NTHREADS)
void retention_fwd_wmma(const unsigned short* __restrict__ wsQ,
                        const unsigned short* __restrict__ wsK,
                        const unsigned short* __restrict__ wsVT,
                        float* __restrict__ Out) {
    __shared__ unsigned short ks [2][BK    * LSTR];
    __shared__ unsigned short vts[2][DHEAD * LSTR];
    __shared__ unsigned short scs[4 * 16 * LSTR];

    const int tid  = threadIdx.x;
    const int lane = tid & 31;
    const int w    = tid >> 5;
    const int ln   = lane & 15;
    const int hi   = lane >> 4;

    const int qb = blockIdx.x & 31;
    const int h  = (blockIdx.x >> 5) & 15;
    const int b  = blockIdx.x >> 9;
    const int i0 = qb * BM;

    const float l2g = head_log2gamma(h);
    const float gamma = exp2f(l2g);
    const float g64   = exp2f(l2g * 64.0f);

    const size_t bh_off = (size_t)b * S_LEN * HID + (size_t)h * DHEAD;
    const unsigned short* gK  = wsK + bh_off;
    const unsigned short* gVT = wsVT + ((size_t)(b * NHEADS + h) * DHEAD) * S_LEN;

    // per-wave Q A-fragments straight from global bf16
    const int qbase = i0 + w * 16;
    Frag16 aq[2];
    {
        const unsigned short* gq = wsQ + bh_off + (size_t)(qbase + ln) * HID;
#pragma unroll
        for (int hf = 0; hf < 2; ++hf) {
            aq[hf].q[0] = *(const uint4*)(gq + hf * 32 + hi * 8);
            aq[hf].q[1] = *(const uint4*)(gq + hf * 32 + 16 + hi * 8);
        }
    }

    v8f acc[4];
#pragma unroll
    for (int i = 0; i < 4; ++i) acc[i] = {};

    // per-row decay factors fv[v] = gamma^{n - jb*64}; updated by *= gamma^64
    float fv[8];
    {
        float g = exp2f(l2g * (float)(w * 16 + 8 * hi));
#pragma unroll
        for (int v2 = 0; v2 < 8; ++v2) { fv[v2] = g; g *= gamma; }
    }

    auto alive = [&](int jb) -> bool {
        int dmin = (qb - jb) * BM - (BK - 1);
        return !(dmin > 0 && (float)dmin * l2g < -40.0f);
    };
    auto stage = [&](int buf, int jb) {
#pragma unroll
        for (int it = 0; it < 4; ++it) {
            int e = tid + it * NTHREADS;
            int r = e >> 3;
            int c = (e & 7) << 3;
            async_b128((unsigned)(uintptr_t)&ks[buf][r * LSTR + c],
                       (unsigned long long)(uintptr_t)(gK + (size_t)(jb * BK + r) * HID + c));
            async_b128((unsigned)(uintptr_t)&vts[buf][r * LSTR + c],
                       (unsigned long long)(uintptr_t)(gVT + (size_t)r * S_LEN + jb * BK + c));
        }
    };

    unsigned short* sw = &scs[w * 16 * LSTR];
    int cur = 0;
    stage(0, qb);
    wait_async0();
    __syncthreads();

    for (int jb = qb; jb >= 0; --jb) {
        const int  njb = jb - 1;
        const bool nxt = (njb >= 0) && alive(njb);
        if (nxt) stage(cur ^ 1, njb);

        if (jb == qb)
            tile_step<true >(ks[cur], vts[cur], sw, aq, acc, fv, w, ln, hi);
        else
            tile_step<false>(ks[cur], vts[cur], sw, aq, acc, fv, w, ln, hi);

        if (!nxt) break;
        wait_async0();
        __syncthreads();
        cur ^= 1;
#pragma unroll
        for (int v2 = 0; v2 < 8; ++v2) fv[v2] *= g64;
    }

    // writeback (C layout: M = v2 + 8*hi, N = ln)
#pragma unroll
    for (int dt = 0; dt < 4; ++dt) {
#pragma unroll
        for (int v2 = 0; v2 < 8; ++v2) {
            int n = qbase + v2 + 8 * hi;
            Out[bh_off + (size_t)n * HID + dt * 16 + ln] = acc[dt][v2];
        }
    }
}

extern "C" void kernel_launch(void* const* d_in, const int* in_sizes, int n_in,
                              void* d_out, int out_size, void* d_ws, size_t ws_size,
                              hipStream_t stream) {
    const float* Q = (const float*)d_in[0];
    const float* K = (const float*)d_in[1];
    const float* V = (const float*)d_in[2];
    float* O = (float*)d_out;
    int Bn = in_sizes[0] / (S_LEN * HID);

    unsigned short* wsQ  = (unsigned short*)d_ws;
    unsigned short* wsK  = wsQ + (size_t)Bn * S_LEN * HID;
    unsigned short* wsVT = wsK + (size_t)Bn * S_LEN * HID;

    dim3 grid((unsigned)(Bn * NHEADS * (S_LEN / BM)));
    retention_prep<<<grid, NTHREADS, 0, stream>>>(Q, K, V, wsQ, wsK, wsVT);
    retention_fwd_wmma<<<grid, NTHREADS, 0, stream>>>(wsQ, wsK, wsVT, O);
}
